// MAGNOEncoder_83897891160641
// MI455X (gfx1250) — matrix operand order repaired and checked by reading, not compiled, not run on hardware
//
#include <hip/hip_runtime.h>
#include <hip/hip_bf16.h>

typedef __attribute__((ext_vector_type(16))) _Float16 v16h;
typedef __attribute__((ext_vector_type(8)))  float    v8f;

#define HID    256
#define TILE_E 128   // edges per block (8 row-tiles of 16)
#define H1S    264   // padded f16 row stride for LDS activation tiles

// Branch-free tanh-gelu: gelu(x) = x - x * rcp(exp2(k*(x+0.044715x^3)) + 1)
// k = 2*log2(e)*sqrt(2/pi). Saturates correctly at +/-inf via v_exp/v_rcp.
__device__ __forceinline__ float gelu_f(float x) {
    float u = x + 0.044715f * x * x * x;
    float z = __builtin_amdgcn_exp2f((2.0f * 1.4426950408889634f * 0.7978845608028654f) * u);
    return x - x * __builtin_amdgcn_rcpf(z + 1.0f);
}

__device__ __forceinline__ void atomic_add_f32(float* p, float v) {
    // non-returning HW fp32 atomic (STOREcnt; s_endpgm waits implicitly)
    asm volatile("global_atomic_add_f32 %0, %1, off" :: "v"(p), "v"(v) : "memory");
}

// Stage 64K f16 (128KB) of fragment-packed weights into LDS with async DMA.
// All 256 threads issue 32 b128 copies each; caller waits + barriers after.
__device__ __forceinline__ void stage_weights_async(const _Float16* __restrict__ Wp,
                                                    _Float16* s_w, int t) {
    const char* g = (const char*)Wp;
    unsigned lbase = (unsigned)(uintptr_t)s_w;   // low 32 bits = LDS address
#pragma unroll
    for (int i = 0; i < 32; ++i) {
        int off = (t + i * 256) * 16;            // 16B chunks, fully coalesced
        unsigned laddr = lbase + (unsigned)off;
        const char* gaddr = g + off;
        asm volatile("global_load_async_to_lds_b128 %0, %1, off"
                     :: "v"(laddr), "v"(gaddr) : "memory");
    }
}

__device__ __forceinline__ void wait_async0() {
    asm volatile("s_wait_asynccnt 0x0" ::: "memory");
}

// A fragment (16-bit A 16x32 ISA layout) from LDS activations, row stride H1S.
__device__ __forceinline__ v16h load_afrag(const _Float16* sh, int lane, int rt, int kt) {
    union { v16h h; unsigned u[8]; } a;
    const int mrow = rt * 16 + (lane & 15);
    const int kA = kt * 32 + ((lane >> 4) << 3);
#pragma unroll
    for (int v = 0; v < 8; ++v) {
        int k = kA + ((v & 4) << 2) + ((v & 3) << 1);
        a.u[v] = *reinterpret_cast<const unsigned*>(&sh[mrow * H1S + k]);
    }
    return a.h;
}

// B fragment: packed contiguous 32B per (ct,kt,lane) in the LDS weight stage.
__device__ __forceinline__ v16h load_bfrag(const _Float16* sw, int lane, int ct, int kt) {
    union { v16h h; float4 q[2]; } b;
    const float4* p = reinterpret_cast<const float4*>(sw + ((((ct * 8 + kt) * 32) + lane) << 4));
    b.q[0] = p[0];
    b.q[1] = p[1];
    return b.h;
}

__device__ __forceinline__ v8f wmma_step(v16h a, v16h b, v8f c) {
    return __builtin_amdgcn_wmma_f32_16x16x32_f16(false, a, false, b, (short)0, c, false, false);
}

// -------- weight prep: f32 row-major -> f16 fragment-packed (B layout) --------
// packed index p: e=p&15, lane=(p>>4)&31, kt=(p>>9)&7, ct=p>>12
// value = W[k][n], n = ct*16+(lane&15), k = kt*32+((lane>>4)<<3)+(e&7)+((e&8)<<1)
__global__ __launch_bounds__(256) void magno_prep_weights(
    const float* __restrict__ W2, const float* __restrict__ W3,
    _Float16* __restrict__ W2p, _Float16* __restrict__ W3p) {
    int idx = blockIdx.x * 256 + threadIdx.x;          // 0 .. 131071
    const float* src = (idx < 65536) ? W2 : W3;
    _Float16*    dst = (idx < 65536) ? W2p : W3p;
    int p = idx & 65535;
    int e = p & 15, lane = (p >> 4) & 31, kt = (p >> 9) & 7, ct = p >> 12;
    int n = ct * 16 + (lane & 15);
    int k = kt * 32 + ((lane >> 4) << 3) + (e & 7) + ((e & 8) << 1);
    dst[p] = (_Float16)src[k * HID + n];
}

__global__ __launch_bounds__(256) void magno_zero(float* __restrict__ buf, int n) {
    int idx = blockIdx.x * 256 + threadIdx.x;
    if (idx < n) buf[idx] = 0.0f;
}

// ---------------- main fused edge-MLP + scatter-add ----------------
__global__ __launch_bounds__(256) void magno_main(
    const float* __restrict__ phys_feats,
    const float* __restrict__ phys_pos,
    const float* __restrict__ latent_pos,
    const int*   __restrict__ edge_src,
    const int*   __restrict__ edge_dst,
    const float* __restrict__ W1,
    const float* __restrict__ b1,
    const float* __restrict__ b2,
    const float* __restrict__ b3,
    const _Float16* __restrict__ W2p,
    const _Float16* __restrict__ W3p,
    float* __restrict__ sums,
    float* __restrict__ counts,
    int E) {
    __shared__ __align__(16) _Float16 s_w[65536];           // 128KB weight stage
    __shared__ __align__(16) _Float16 s_h1[TILE_E * H1S];   // 66KB
    __shared__ __align__(16) _Float16 s_h2[TILE_E * H1S];   // 66KB
    __shared__ __align__(16) float    s_ein[TILE_E * 12];   // 6KB
    __shared__ int                    s_dst[TILE_E];

    const int t    = threadIdx.x;
    const int lane = t & 31;
    const int w    = t >> 5;
    const int g    = w >> 1;            // row-tile group: rts {2g, 2g+1} of 8
    const int cg   = (w & 1) * 8;       // col-tile base: 8 cts of 16
    const long e0  = (long)blockIdx.x * TILE_E;

    // ---- kick off async staging of layer-2 weights (overlaps gather+layer1) ----
    stage_weights_async(W2p, s_w, t);

    // ---- gather edge inputs ----
    if (t < TILE_E) {
        long e = e0 + t;
        if (e < E) {
            int s = edge_src[e];
            int d = edge_dst[e];
            s_dst[t] = d;
            float f0 = phys_feats[s * 3 + 0], f1 = phys_feats[s * 3 + 1], f2 = phys_feats[s * 3 + 2];
            float p0 = phys_pos[s * 3 + 0],   p1 = phys_pos[s * 3 + 1],   p2 = phys_pos[s * 3 + 2];
            float l0 = latent_pos[d * 3 + 0], l1 = latent_pos[d * 3 + 1], l2 = latent_pos[d * 3 + 2];
            s_ein[t * 12 + 0] = f0; s_ein[t * 12 + 1] = f1; s_ein[t * 12 + 2] = f2;
            s_ein[t * 12 + 3] = p0; s_ein[t * 12 + 4] = p1; s_ein[t * 12 + 5] = p2;
            s_ein[t * 12 + 6] = l0 - p0; s_ein[t * 12 + 7] = l1 - p1; s_ein[t * 12 + 8] = l2 - p2;
            atomic_add_f32(&counts[d], 1.0f);
        } else {
            s_dst[t] = -1;
#pragma unroll
            for (int i = 0; i < 9; ++i) s_ein[t * 12 + i] = 0.0f;
        }
    }
    __syncthreads();

    // ---- layer 1 (K=9, scalar VALU): thread t owns hidden column c = t ----
    {
        float w1r[9];
#pragma unroll
        for (int i = 0; i < 9; ++i) w1r[i] = W1[i * HID + t];
        float bias = b1[t];
        for (int e = 0; e < TILE_E; ++e) {
            float acc = bias;
#pragma unroll
            for (int i = 0; i < 9; ++i) acc += s_ein[e * 12 + i] * w1r[i];
            s_h1[e * H1S + t] = (_Float16)gelu_f(acc);
        }
    }
    wait_async0();        // W2 staged
    __syncthreads();      // h1 ready, all waves' DMA done

    // ---- layer 2: h2 = gelu(h1 @ W2 + b2); RT=2 x CT=8 per wave ----
    {
        v16h a0[8], a1[8];
#pragma unroll
        for (int kt = 0; kt < 8; ++kt) {
            a0[kt] = load_afrag(s_h1, lane, 2 * g + 0, kt);
            a1[kt] = load_afrag(s_h1, lane, 2 * g + 1, kt);
        }
        for (int ci = 0; ci < 8; ++ci) {
            int ct = cg + ci;
            int ncol = ct * 16 + (lane & 15);
            float bias = b2[ncol];
            v8f acc0, acc1;
#pragma unroll
            for (int r = 0; r < 8; ++r) { acc0[r] = bias; acc1[r] = bias; }
#pragma unroll
            for (int kt = 0; kt < 8; ++kt) {
                v16h bf = load_bfrag(s_w, lane, ct, kt);
                acc0 = wmma_step(a0[kt], bf, acc0);
                acc1 = wmma_step(a1[kt], bf, acc1);
            }
#pragma unroll
            for (int r = 0; r < 8; ++r) {
                int mr = ((lane >> 4) << 3) + r;
                s_h2[((2 * g + 0) * 16 + mr) * H1S + ncol] = (_Float16)gelu_f(acc0[r]);
                s_h2[((2 * g + 1) * 16 + mr) * H1S + ncol] = (_Float16)gelu_f(acc1[r]);
            }
        }
    }
    __syncthreads();      // everyone done with s_w(W2) and h1; h2 ready

    // ---- stage layer-3 weights ----
    stage_weights_async(W3p, s_w, t);
    wait_async0();
    __syncthreads();

    // ---- layer 3: msg = h2 @ W3 + b3, scatter-add into L2-resident sums ----
    {
        v16h a0[8], a1[8];
#pragma unroll
        for (int kt = 0; kt < 8; ++kt) {
            a0[kt] = load_afrag(s_h2, lane, 2 * g + 0, kt);
            a1[kt] = load_afrag(s_h2, lane, 2 * g + 1, kt);
        }
        for (int ci = 0; ci < 8; ++ci) {
            int ct = cg + ci;
            int ncol = ct * 16 + (lane & 15);
            float bias = b3[ncol];
            v8f acc0, acc1;
#pragma unroll
            for (int r = 0; r < 8; ++r) { acc0[r] = bias; acc1[r] = bias; }
#pragma unroll
            for (int kt = 0; kt < 8; ++kt) {
                v16h bf = load_bfrag(s_w, lane, ct, kt);
                acc0 = wmma_step(a0[kt], bf, acc0);
                acc1 = wmma_step(a1[kt], bf, acc1);
            }
#pragma unroll
            for (int r = 0; r < 8; ++r) {
                int mr = ((lane >> 4) << 3) + r;
                int m0 = (2 * g + 0) * 16 + mr;
                int m1 = (2 * g + 1) * 16 + mr;
                int d0 = s_dst[m0];
                int d1 = s_dst[m1];
                if (d0 >= 0) atomic_add_f32(&sums[(size_t)d0 * HID + ncol], acc0[r]);
                if (d1 >= 0) atomic_add_f32(&sums[(size_t)d1 * HID + ncol], acc1[r]);
            }
        }
    }
}

__global__ __launch_bounds__(256) void magno_finalize(
    const float* __restrict__ sums, const float* __restrict__ counts,
    float* __restrict__ out, int n) {
    int idx = blockIdx.x * 256 + threadIdx.x;
    if (idx < n) {
        float c = counts[idx >> 8];
        out[idx] = sums[idx] / fmaxf(c, 1.0f);
    }
}

extern "C" void kernel_launch(void* const* d_in, const int* in_sizes, int n_in,
                              void* d_out, int out_size, void* d_ws, size_t ws_size,
                              hipStream_t stream) {
    const float* phys_feats = (const float*)d_in[0];
    const float* phys_pos   = (const float*)d_in[1];
    const float* latent_pos = (const float*)d_in[2];
    const int*   edge_src   = (const int*)d_in[3];
    const int*   edge_dst   = (const int*)d_in[4];
    const float* W1 = (const float*)d_in[5];
    const float* b1 = (const float*)d_in[6];
    const float* W2 = (const float*)d_in[7];
    const float* b2 = (const float*)d_in[8];
    const float* W3 = (const float*)d_in[9];
    const float* b3 = (const float*)d_in[10];

    const int E = in_sizes[3];              // 1,048,576
    const int N_LATENT = out_size / HID;    // 4096

    // workspace layout
    char* ws = (char*)d_ws;
    _Float16* W2p    = (_Float16*)ws;                              // 128 KB
    _Float16* W3p    = (_Float16*)(ws + 131072);                   // 128 KB
    float*    sums   = (float*)(ws + 262144);                      // 4 MB
    float*    counts = (float*)(ws + 262144 + (size_t)N_LATENT * HID * 4);

    magno_prep_weights<<<512, 256, 0, stream>>>(W2, W3, W2p, W3p);

    int nzero = N_LATENT * HID + N_LATENT;  // sums + counts contiguous
    magno_zero<<<(nzero + 255) / 256, 256, 0, stream>>>(sums, nzero);

    int nblocks = (E + TILE_E - 1) / TILE_E;
    magno_main<<<nblocks, 256, 0, stream>>>(
        phys_feats, phys_pos, latent_pos, edge_src, edge_dst,
        W1, b1, b2, b3, W2p, W3p, sums, counts, E);

    magno_finalize<<<(out_size + 255) / 256, 256, 0, stream>>>(
        sums, counts, (float*)d_out, out_size);
}
